// MinimalDecoderLayer_62792421868198
// MI455X (gfx1250) — compile-verified, matrix operand order checked by reading, not run on hardware
//
#include <hip/hip_runtime.h>
#include <math.h>

// ---------------------------------------------------------------------------
// MLA decoder layer for gfx1250 (MI455X).
//  - All weights pre-converted fp32->f16 once per launch; all GEMM activations
//    produced as f16 twins by their producer kernels.
//  - GEMMs: 128x128 tile, double-buffered LDS staged with
//    GLOBAL_LOAD_ASYNC_TO_LDS_B128 (ASYNCcnt / s_wait_asynccnt), f16 operands,
//    v_wmma_f32_16x16x32_f16 with fp32 accumulation.
//  - Attention: per-wave flash attention, 20 WMMAs per 32-key chunk.
// Fragment layouts per CDNA5 ISA 7.12.2:
//   A (16x32 f16): lane l holds row m=l&15; halfs 0..7 = K 8*hi..8*hi+7,
//                  halfs 8..15 = K 16+8*hi..+7   (hi = l>>4)
//   B (32x16 f16): lane l holds col n=l&15; same K packing as A
//   C/D (16x16 f32): lane l holds col n=l&15, VGPR r holds row m=r+8*hi
// ---------------------------------------------------------------------------

#define S_LEN   2048
#define H_DIM   2048
#define QRANK   768
#define KVRANK  512
#define ROPE_DD 64
#define NOPE_DD 128
#define VD      128
#define NHEADS  16
#define QKD     (NOPE_DD + ROPE_DD)   // 192
#define NEXP    8
#define IMOE    1024
#define IDENSE  8192
#define RMS_EPS 1e-6f

typedef __attribute__((ext_vector_type(16))) _Float16 v16h;
typedef __attribute__((ext_vector_type(8)))  _Float16 v8h;
typedef __attribute__((ext_vector_type(4)))  _Float16 v4h;
typedef __attribute__((ext_vector_type(8)))  float    v8f;

union HF16 { v16h v; _Float16 e[16]; };
union F8   { v8f  v; float    e[8];  };

__device__ __forceinline__ v16h ld_frag_lds(const _Float16* rowPtr, int hi) {
  v8h lo = *reinterpret_cast<const v8h*>(rowPtr + 8 * hi);
  v8h hh = *reinterpret_cast<const v8h*>(rowPtr + 16 + 8 * hi);
  return __builtin_shufflevector(lo, hh, 0,1,2,3,4,5,6,7,8,9,10,11,12,13,14,15);
}

// ---------------------------------------------------------------------------
// fp32 -> f16 bulk conversion (weights), 4 elements/thread, n % 4 == 0.
// ---------------------------------------------------------------------------
__global__ __launch_bounds__(256) void f32_to_f16_kernel(
    const float* __restrict__ src, _Float16* __restrict__ dst, int n4)
{
  int i = blockIdx.x * 256 + threadIdx.x;
  if (i >= n4) return;
  float4 v = reinterpret_cast<const float4*>(src)[i];
  v4h h;
  h[0] = (_Float16)v.x; h[1] = (_Float16)v.y;
  h[2] = (_Float16)v.z; h[3] = (_Float16)v.w;
  reinterpret_cast<v4h*>(dst)[i] = h;
}

// ---------------------------------------------------------------------------
// GEMM: C[M,N] (+)= A[M,K] * B[N,K]^T with f16 A/B (K-contiguous rows).
// Requires M%128==0, K%32==0; ragged N (kv_a: 576) handled by clamped B rows
// + column-guarded stores, so EXEC is all-ones around every WMMA.
// 256 threads (8 waves), tile 128x128, K-step 32, double-buffered 32 KB LDS
// staged via GLOBAL_LOAD_ASYNC_TO_LDS_B128.  Wave grid 4(M)x2(N): each wave
// computes 32x64 = 2 A-frags x 4 B-frags = 8 WMMAs per K-step.
// ---------------------------------------------------------------------------
__global__ __launch_bounds__(256) void gemm_xwt_wmma(
    const _Float16* __restrict__ A, const _Float16* __restrict__ B,
    float* __restrict__ C, int M, int N, int K, int accumulate)
{
  __shared__ __align__(32) _Float16 As[2][128][32];   // 16 KB
  __shared__ __align__(32) _Float16 Bs[2][128][32];   // 16 KB
  const int tid  = threadIdx.x;
  const int wave = tid >> 5;
  const int lane = tid & 31;
  const int fm   = lane & 15;
  const int hi   = lane >> 4;
  const int m0   = blockIdx.y << 7;
  const int n0   = blockIdx.x << 7;
  const int wm   = wave & 3;            // 32-row M band
  const int wn   = wave >> 2;           // 64-col N band

  const int lrow  = tid >> 1;           // 0..127
  const int lcolh = (tid & 1) << 4;     // half index: 0 or 16

  int brow = n0 + lrow; if (brow > N - 1) brow = N - 1;   // clamp; junk discarded
  const _Float16* aPtr = A + (size_t)(m0 + lrow) * K + lcolh;
  const _Float16* bPtr = B + (size_t)brow * K + lcolh;

  // LDS aperture sits in addr[63:32]; low 32 bits of a flat shared pointer are
  // exactly the LDS byte offset the async VDST expects (ISA 10.2).
  unsigned aLds[2], bLds[2];
  aLds[0] = (unsigned)(size_t)&As[0][lrow][lcolh];
  aLds[1] = (unsigned)(size_t)&As[1][lrow][lcolh];
  bLds[0] = (unsigned)(size_t)&Bs[0][lrow][lcolh];
  bLds[1] = (unsigned)(size_t)&Bs[1][lrow][lcolh];

  auto stage = [&](int buf, int kOff) {
    unsigned long long ga = (unsigned long long)(const void*)(aPtr + kOff);
    unsigned long long gb = (unsigned long long)(const void*)(bPtr + kOff);
    // 32 bytes per thread per operand; offset: applies to LDS and global alike.
    asm volatile("global_load_async_to_lds_b128 %0, %1, off"
                 :: "v"(aLds[buf]), "v"(ga) : "memory");
    asm volatile("global_load_async_to_lds_b128 %0, %1, off offset:16"
                 :: "v"(aLds[buf]), "v"(ga) : "memory");
    asm volatile("global_load_async_to_lds_b128 %0, %1, off"
                 :: "v"(bLds[buf]), "v"(gb) : "memory");
    asm volatile("global_load_async_to_lds_b128 %0, %1, off offset:16"
                 :: "v"(bLds[buf]), "v"(gb) : "memory");
  };

  v8f vzero = {};
  v8f acc[2][4];
  #pragma unroll
  for (int i = 0; i < 2; ++i)
    #pragma unroll
    for (int j = 0; j < 4; ++j) acc[i][j] = vzero;

  const int nsteps = K >> 5;
  stage(0, 0);
  for (int s = 0; s < nsteps; ++s) {
    const int cur = s & 1, nxt = cur ^ 1;
    __syncthreads();                       // all waves done reading buf[nxt]
    if (s + 1 < nsteps) {
      stage(nxt, (s + 1) << 5);            // prefetch next tile
      asm volatile("s_wait_asynccnt 0x4" ::: "memory");  // cur's 4 landed
    } else {
      asm volatile("s_wait_asynccnt 0x0" ::: "memory");
    }
    __syncthreads();                       // every lane's cur data visible

    v16h af0 = ld_frag_lds(&As[cur][wm * 32 + fm][0], hi);
    v16h af1 = ld_frag_lds(&As[cur][wm * 32 + 16 + fm][0], hi);
    #pragma unroll
    for (int j = 0; j < 4; ++j) {
      v16h bf = ld_frag_lds(&Bs[cur][wn * 64 + j * 16 + fm][0], hi);
      acc[0][j] = __builtin_amdgcn_wmma_f32_16x16x32_f16(false, af0, false, bf,
                                                         (short)0, acc[0][j],
                                                         false, false);
      acc[1][j] = __builtin_amdgcn_wmma_f32_16x16x32_f16(false, af1, false, bf,
                                                         (short)0, acc[1][j],
                                                         false, false);
    }
  }

  if (accumulate) {
    #pragma unroll
    for (int i = 0; i < 2; ++i)
      #pragma unroll
      for (int j = 0; j < 4; ++j) {
        F8 rr; rr.v = acc[i][j];
        int col = n0 + wn * 64 + j * 16 + fm;
        if (col < N) {
          #pragma unroll
          for (int r = 0; r < 8; ++r) {
            size_t idx = (size_t)(m0 + wm * 32 + i * 16 + r + 8 * hi) * N + col;
            C[idx] = C[idx] + rr.e[r];
          }
        }
      }
  } else {
    #pragma unroll
    for (int i = 0; i < 2; ++i)
      #pragma unroll
      for (int j = 0; j < 4; ++j) {
        F8 rr; rr.v = acc[i][j];
        int col = n0 + wn * 64 + j * 16 + fm;
        if (col < N) {
          #pragma unroll
          for (int r = 0; r < 8; ++r) {
            size_t idx = (size_t)(m0 + wm * 32 + i * 16 + r + 8 * hi) * N + col;
            C[idx] = rr.e[r];
          }
        }
      }
  }
}

// ---------------------------------------------------------------------------
// Flash attention, one wave per (head, 16 query rows).  Q pre-scaled by
// 192^-0.5 at fragment load.  K chunk = 32 keys per iteration.  Output f16
// (feeds the o-proj GEMM directly).
// ---------------------------------------------------------------------------
__global__ __launch_bounds__(32) void mla_flash_attn_kernel(
    const float* __restrict__ q,    // [S, NHEADS*192]  (RoPE applied)
    const float* __restrict__ kve,  // [S, NHEADS*256]  (k_nope | v)
    const float* __restrict__ kpe,  // [S, 64]          (roped, shared by heads)
    _Float16* __restrict__ outb)    // [S, NHEADS*128]
{
  __shared__ _Float16 sP[16][32];   // C-layout -> A-layout transpose staging
  const int lane = threadIdx.x;
  const int fm   = lane & 15;
  const int hi   = lane >> 4;
  const int head = blockIdx.x & (NHEADS - 1);
  const int m0   = (blockIdx.x >> 4) * 16;
  const float qscale = rsqrtf((float)QKD);

  HF16 qf[6];
  {
    const float* qrow = q + (size_t)(m0 + fm) * (NHEADS * QKD) + head * QKD;
    #pragma unroll
    for (int kk = 0; kk < 6; ++kk) {
      #pragma unroll
      for (int j = 0; j < 8; ++j) {
        qf[kk].e[j]     = (_Float16)(qrow[kk * 32 + 8 * hi + j] * qscale);
        qf[kk].e[j + 8] = (_Float16)(qrow[kk * 32 + 16 + 8 * hi + j] * qscale);
      }
    }
  }

  F8 oacc[8];
  float rmax[8], rsum[8];
  #pragma unroll
  for (int r = 0; r < 8; ++r) { rmax[r] = -1e30f; rsum[r] = 0.f; }
  #pragma unroll
  for (int n = 0; n < 8; ++n)
    #pragma unroll
    for (int r = 0; r < 8; ++r) oacc[n].e[r] = 0.f;

  for (int t0 = 0; t0 < S_LEN; t0 += 32) {
    F8 s0, s1;
    #pragma unroll
    for (int r = 0; r < 8; ++r) { s0.e[r] = 0.f; s1.e[r] = 0.f; }
    {
      const int tA = t0 + fm, tB = t0 + 16 + fm;
      const float* kA = kve + (size_t)tA * (NHEADS * 256) + head * 256;
      const float* kB = kve + (size_t)tB * (NHEADS * 256) + head * 256;
      const float* pA = kpe + (size_t)tA * ROPE_DD;
      const float* pB = kpe + (size_t)tB * ROPE_DD;
      #pragma unroll
      for (int kk = 0; kk < 6; ++kk) {
        HF16 b0, b1;
        #pragma unroll
        for (int j = 0; j < 8; ++j) {
          int kd0 = kk * 32 + 8 * hi + j;
          int kd1 = kk * 32 + 16 + 8 * hi + j;
          b0.e[j]     = (_Float16)(kd0 < NOPE_DD ? kA[kd0] : pA[kd0 - NOPE_DD]);
          b0.e[j + 8] = (_Float16)(kd1 < NOPE_DD ? kA[kd1] : pA[kd1 - NOPE_DD]);
          b1.e[j]     = (_Float16)(kd0 < NOPE_DD ? kB[kd0] : pB[kd0 - NOPE_DD]);
          b1.e[j + 8] = (_Float16)(kd1 < NOPE_DD ? kB[kd1] : pB[kd1 - NOPE_DD]);
        }
        s0.v = __builtin_amdgcn_wmma_f32_16x16x32_f16(false, qf[kk].v, false, b0.v,
                                                      (short)0, s0.v, false, false);
        s1.v = __builtin_amdgcn_wmma_f32_16x16x32_f16(false, qf[kk].v, false, b1.v,
                                                      (short)0, s1.v, false, false);
      }
    }

    float corr[8];
    #pragma unroll
    for (int r = 0; r < 8; ++r) {
      float mx = fmaxf(s0.e[r], s1.e[r]);
      mx = fmaxf(mx, __shfl_xor(mx, 1));
      mx = fmaxf(mx, __shfl_xor(mx, 2));
      mx = fmaxf(mx, __shfl_xor(mx, 4));
      mx = fmaxf(mx, __shfl_xor(mx, 8));
      float mnew = fmaxf(rmax[r], mx);
      float c  = __expf(rmax[r] - mnew);
      float p0 = __expf(s0.e[r] - mnew);
      float p1 = __expf(s1.e[r] - mnew);
      float ps = p0 + p1;
      ps += __shfl_xor(ps, 1);
      ps += __shfl_xor(ps, 2);
      ps += __shfl_xor(ps, 4);
      ps += __shfl_xor(ps, 8);
      rsum[r] = rsum[r] * c + ps;
      rmax[r] = mnew;
      corr[r] = c;
      s0.e[r] = p0; s1.e[r] = p1;
    }
    #pragma unroll
    for (int n = 0; n < 8; ++n)
      #pragma unroll
      for (int r = 0; r < 8; ++r) oacc[n].e[r] *= corr[r];

    __syncthreads();
    #pragma unroll
    for (int r = 0; r < 8; ++r) {
      sP[r + 8 * hi][fm]      = (_Float16)s0.e[r];
      sP[r + 8 * hi][16 + fm] = (_Float16)s1.e[r];
    }
    __syncthreads();
    v16h pf = ld_frag_lds(&sP[fm][0], hi);

    #pragma unroll
    for (int n = 0; n < 8; ++n) {
      HF16 bf;
      #pragma unroll
      for (int j = 0; j < 8; ++j) {
        int k0 = 8 * hi + j, k1 = 16 + 8 * hi + j;
        bf.e[j]     = (_Float16)kve[(size_t)(t0 + k0) * (NHEADS * 256)
                                    + head * 256 + NOPE_DD + n * 16 + fm];
        bf.e[j + 8] = (_Float16)kve[(size_t)(t0 + k1) * (NHEADS * 256)
                                    + head * 256 + NOPE_DD + n * 16 + fm];
      }
      oacc[n].v = __builtin_amdgcn_wmma_f32_16x16x32_f16(false, pf, false, bf.v,
                                                         (short)0, oacc[n].v,
                                                         false, false);
    }
  }

  #pragma unroll
  for (int n = 0; n < 8; ++n)
    #pragma unroll
    for (int r = 0; r < 8; ++r)
      outb[(size_t)(m0 + r + 8 * hi) * (NHEADS * VD) + head * VD + n * 16 + fm]
          = (_Float16)(oacc[n].e[r] / rsum[r]);
}

// ---------------------------------------------------------------------------
// Elementwise / reduction helpers (fp32 math; f16 twins for GEMM operands)
// ---------------------------------------------------------------------------
__global__ __launch_bounds__(256) void rmsnorm_kernel(
    const float* __restrict__ in, int istride, const float* __restrict__ w,
    _Float16* __restrict__ out16, int ostride, int dim)
{
  __shared__ float red[256];
  const int tid = threadIdx.x;
  const float* x = in + (size_t)blockIdx.x * istride;
  _Float16* y = out16 + (size_t)blockIdx.x * ostride;
  float s = 0.f;
  for (int j = tid; j < dim; j += 256) { float v = x[j]; s += v * v; }
  red[tid] = s; __syncthreads();
  for (int k = 128; k > 0; k >>= 1) {
    if (tid < k) red[tid] += red[tid + k];
    __syncthreads();
  }
  float inv = rsqrtf(red[0] / (float)dim + RMS_EPS);
  for (int j = tid; j < dim; j += 256) y[j] = (_Float16)(w[j] * x[j] * inv);
}

__global__ __launch_bounds__(256) void add_rmsnorm2048_kernel(
    const float* __restrict__ x, const float* __restrict__ y,
    const float* __restrict__ w, float* __restrict__ out,
    _Float16* __restrict__ out16)
{
  __shared__ float red[256];
  const int tid = threadIdx.x;
  const size_t base = (size_t)blockIdx.x * 2048;
  float v[8]; float s = 0.f;
  #pragma unroll
  for (int i = 0; i < 8; ++i) {
    int j = tid + i * 256;
    v[i] = x[base + j] + y[base + j];
    s += v[i] * v[i];
  }
  red[tid] = s; __syncthreads();
  for (int k = 128; k > 0; k >>= 1) {
    if (tid < k) red[tid] += red[tid + k];
    __syncthreads();
  }
  float inv = rsqrtf(red[0] / 2048.0f + RMS_EPS);
  #pragma unroll
  for (int i = 0; i < 8; ++i) {
    int j = tid + i * 256;
    float r = w[j] * v[i] * inv;
    out[base + j] = r;
    if (out16) out16[base + j] = (_Float16)r;
  }
}

__global__ __launch_bounds__(256) void rope_q_kernel(float* __restrict__ q)
{
  int idx = blockIdx.x * 256 + threadIdx.x;   // S * NHEADS * 32 threads
  int p = idx & 31;
  int pid = idx >> 5;
  int h = pid & (NHEADS - 1);
  int t = pid >> 4;
  float* x = q + (size_t)t * (NHEADS * QKD) + h * QKD + NOPE_DD;
  float invf = __powf(10000.0f, -(float)p / 32.0f);
  float f = (float)t * invf;
  float c = __cosf(f), s = __sinf(f);
  float x0 = x[p], x1 = x[p + 32];
  x[p]      = x0 * c - x1 * s;
  x[p + 32] = x1 * c + x0 * s;
}

__global__ __launch_bounds__(256) void rope_kpe_kernel(
    const float* __restrict__ kva, float* __restrict__ kpe)
{
  int idx = blockIdx.x * 256 + threadIdx.x;   // S * 32 threads
  int p = idx & 31;
  int t = idx >> 5;
  const float* x = kva + (size_t)t * (KVRANK + ROPE_DD) + KVRANK;
  float invf = __powf(10000.0f, -(float)p / 32.0f);
  float f = (float)t * invf;
  float c = __cosf(f), s = __sinf(f);
  float x0 = x[p], x1 = x[p + 32];
  kpe[(size_t)t * ROPE_DD + p]      = x0 * c - x1 * s;
  kpe[(size_t)t * ROPE_DD + p + 32] = x1 * c + x0 * s;
}

__global__ __launch_bounds__(256) void router_kernel(
    const float* __restrict__ hs, const float* __restrict__ rw,
    const float* __restrict__ ebias, float* __restrict__ comb)
{
  __shared__ float logits[NEXP];
  const int tid = threadIdx.x, e = tid >> 5, lane = tid & 31;
  const int t = blockIdx.x;
  const float* x = hs + (size_t)t * H_DIM;
  const float* w = rw + (size_t)e * H_DIM;
  float s = 0.f;
  for (int d = lane; d < H_DIM; d += 32) s += x[d] * w[d];
  s += __shfl_xor(s, 16); s += __shfl_xor(s, 8);
  s += __shfl_xor(s, 4);  s += __shfl_xor(s, 2); s += __shfl_xor(s, 1);
  if (lane == 0) logits[e] = s;
  __syncthreads();
  if (tid == 0) {
    float sc[NEXP];
    #pragma unroll
    for (int i = 0; i < NEXP; ++i)
      sc[i] = 1.f / (1.f + __expf(-logits[i])) + ebias[i];
    int i0 = 0; float v0 = sc[0];
    for (int i = 1; i < NEXP; ++i) if (sc[i] > v0) { v0 = sc[i]; i0 = i; }
    int i1 = -1; float v1 = -1e30f;
    for (int i = 0; i < NEXP; ++i)
      if (i != i0 && sc[i] > v1) { v1 = sc[i]; i1 = i; }
    for (int i = 0; i < NEXP; ++i)
      comb[(size_t)t * NEXP + i] = (i == i0) ? v0 : (i == i1) ? v1 : 0.f;
  }
}

// out16 = f16( silu(g) * u * (scale ? scale[t*ss] : 1) )
__global__ __launch_bounds__(256) void silu_mul_kernel(
    const float* __restrict__ g, int gs, const float* __restrict__ u, int us,
    _Float16* __restrict__ out16, int os, const float* __restrict__ scale,
    int ss, int cols, int total)
{
  int idx = blockIdx.x * 256 + threadIdx.x;
  if (idx >= total) return;
  int t = idx / cols;
  int i = idx - t * cols;
  float gg = g[(size_t)t * gs + i];
  float uu = u[(size_t)t * us + i];
  float r = (gg / (1.f + __expf(-gg))) * uu;
  if (scale) r *= scale[(size_t)t * ss];
  out16[(size_t)t * os + i] = (_Float16)r;
}

// ---------------------------------------------------------------------------
extern "C" void kernel_launch(void* const* d_in, const int* in_sizes, int n_in,
                              void* d_out, int out_size, void* d_ws, size_t ws_size,
                              hipStream_t stream)
{
  (void)in_sizes; (void)n_in; (void)out_size; (void)ws_size;
  const float* hidden     = (const float*)d_in[0];
  const float* in_norm_w  = (const float*)d_in[1];
  const float* q_a_w      = (const float*)d_in[2];
  const float* q_a_norm_w = (const float*)d_in[3];
  const float* q_b_w      = (const float*)d_in[4];
  const float* kv_a_w     = (const float*)d_in[5];
  const float* kv_a_nw    = (const float*)d_in[6];
  const float* kv_b_w     = (const float*)d_in[7];
  const float* o_w        = (const float*)d_in[8];
  const float* pa_norm_w  = (const float*)d_in[9];
  const float* router_w   = (const float*)d_in[10];
  const float* e_bias     = (const float*)d_in[11];
  const float* egu        = (const float*)d_in[12];
  const float* edown      = (const float*)d_in[13];
  const float* sh_gate    = (const float*)d_in[14];
  const float* sh_up      = (const float*)d_in[15];
  const float* sh_down    = (const float*)d_in[16];
  const float* pm_norm_w  = (const float*)d_in[17];
  float* out = (float*)d_out;

  // ---- fp32 workspace ----
  float* W = (float*)d_ws;
  size_t off = 0;
  auto take = [&](size_t n) { float* p = W + off; off += n; return p; };
  float* qa    = take((size_t)S_LEN * QRANK);
  float* qbuf  = take((size_t)S_LEN * NHEADS * QKD);
  float* kva   = take((size_t)S_LEN * (KVRANK + ROPE_DD));
  float* kpe   = take((size_t)S_LEN * ROPE_DD);
  float* kve   = take((size_t)S_LEN * NHEADS * (NOPE_DD + VD));
  float* oproj = take((size_t)S_LEN * H_DIM);
  float* hs2   = take((size_t)S_LEN * H_DIM);
  float* comb  = take((size_t)S_LEN * NEXP);
  float* gu    = take((size_t)S_LEN * 2 * IMOE);
  float* moe   = take((size_t)S_LEN * H_DIM);
  float* ubuf  = take((size_t)S_LEN * IDENSE);
  // shared-gate fp32 buffer aliases the attention-phase region (dead by then):
  // [qa..kve] = 17.56M floats >= 16.78M needed.
  float* gbuf = qa;

  // ---- f16 workspace (weight copies + activation twins) ----
  _Float16* Hh = (_Float16*)(W + off);
  size_t hoff = 0;
  auto takeh = [&](size_t n) { _Float16* p = Hh + hoff; hoff += n; return p; };
  _Float16* w_qa16 = takeh((size_t)QRANK * H_DIM);
  _Float16* w_qb16 = takeh((size_t)NHEADS * QKD * QRANK);
  _Float16* w_kva16 = takeh((size_t)(KVRANK + ROPE_DD) * H_DIM);
  _Float16* w_kvb16 = takeh((size_t)NHEADS * (NOPE_DD + VD) * KVRANK);
  _Float16* w_o16 = takeh((size_t)H_DIM * NHEADS * VD);
  _Float16* w_egu16 = takeh((size_t)NEXP * 2 * IMOE * H_DIM);
  _Float16* w_edown16 = takeh((size_t)NEXP * H_DIM * IMOE);
  _Float16* w_sg16 = takeh((size_t)IDENSE * H_DIM);
  _Float16* w_su16 = takeh((size_t)IDENSE * H_DIM);
  _Float16* w_sd16 = takeh((size_t)H_DIM * IDENSE);
  _Float16* normed16 = takeh((size_t)S_LEN * H_DIM);
  _Float16* qa16   = takeh((size_t)S_LEN * QRANK);
  _Float16* kvc16  = takeh((size_t)S_LEN * KVRANK);
  _Float16* attn16 = takeh((size_t)S_LEN * NHEADS * VD);
  _Float16* hs216  = takeh((size_t)S_LEN * H_DIM);
  _Float16* act16  = takeh((size_t)S_LEN * IMOE);
  _Float16* gact16 = takeh((size_t)S_LEN * IDENSE);

  dim3 blk(256);
  auto cvt = [&](const float* s, _Float16* d, size_t n) {
    int n4 = (int)(n / 4);
    f32_to_f16_kernel<<<(n4 + 255) / 256, blk, 0, stream>>>(s, d, n4);
  };
  auto gemm = [&](const _Float16* A, const _Float16* B, float* C,
                  int M, int N, int K, int acc) {
    dim3 grid((N + 127) / 128, M / 128);
    gemm_xwt_wmma<<<grid, blk, 0, stream>>>(A, B, C, M, N, K, acc);
  };

  // ---- one-shot weight conversion (runs every launch; deterministic) ----
  cvt(q_a_w,  w_qa16,  (size_t)QRANK * H_DIM);
  cvt(q_b_w,  w_qb16,  (size_t)NHEADS * QKD * QRANK);
  cvt(kv_a_w, w_kva16, (size_t)(KVRANK + ROPE_DD) * H_DIM);
  cvt(kv_b_w, w_kvb16, (size_t)NHEADS * (NOPE_DD + VD) * KVRANK);
  cvt(o_w,    w_o16,   (size_t)H_DIM * NHEADS * VD);
  cvt(egu,    w_egu16, (size_t)NEXP * 2 * IMOE * H_DIM);
  cvt(edown,  w_edown16, (size_t)NEXP * H_DIM * IMOE);
  cvt(sh_gate, w_sg16, (size_t)IDENSE * H_DIM);
  cvt(sh_up,   w_su16, (size_t)IDENSE * H_DIM);
  cvt(sh_down, w_sd16, (size_t)H_DIM * IDENSE);

  // ---- attention path ----
  rmsnorm_kernel<<<S_LEN, blk, 0, stream>>>(hidden, H_DIM, in_norm_w,
                                            normed16, H_DIM, H_DIM);
  gemm(normed16, w_qa16, qa, S_LEN, QRANK, H_DIM, 0);
  rmsnorm_kernel<<<S_LEN, blk, 0, stream>>>(qa, QRANK, q_a_norm_w,
                                            qa16, QRANK, QRANK);
  gemm(qa16, w_qb16, qbuf, S_LEN, NHEADS * QKD, QRANK, 0);
  gemm(normed16, w_kva16, kva, S_LEN, KVRANK + ROPE_DD, H_DIM, 0);
  rmsnorm_kernel<<<S_LEN, blk, 0, stream>>>(kva, KVRANK + ROPE_DD, kv_a_nw,
                                            kvc16, KVRANK, KVRANK);
  rope_kpe_kernel<<<(S_LEN * 32) / 256, blk, 0, stream>>>(kva, kpe);
  gemm(kvc16, w_kvb16, kve, S_LEN, NHEADS * (NOPE_DD + VD), KVRANK, 0);
  rope_q_kernel<<<(S_LEN * NHEADS * 32) / 256, blk, 0, stream>>>(qbuf);

  mla_flash_attn_kernel<<<NHEADS * (S_LEN / 16), dim3(32), 0, stream>>>(
      qbuf, kve, kpe, attn16);

  gemm(attn16, w_o16, oproj, S_LEN, H_DIM, NHEADS * VD, 0);
  add_rmsnorm2048_kernel<<<S_LEN, blk, 0, stream>>>(hidden, oproj,
                                                    pa_norm_w, hs2, hs216);

  // ---- MoE (dense over all experts, top-2 combine via scaled act) ----
  router_kernel<<<S_LEN, blk, 0, stream>>>(hs2, router_w, e_bias, comb);
  for (int e = 0; e < NEXP; ++e) {
    const _Float16* wgu = w_egu16 + (size_t)e * 2 * IMOE * H_DIM;
    const _Float16* wdn = w_edown16 + (size_t)e * H_DIM * IMOE;
    gemm(hs216, wgu, gu, S_LEN, 2 * IMOE, H_DIM, 0);
    int tot = S_LEN * IMOE;
    silu_mul_kernel<<<(tot + 255) / 256, blk, 0, stream>>>(
        gu, 2 * IMOE, gu + IMOE, 2 * IMOE, act16, IMOE,
        comb + e, NEXP, IMOE, tot);
    gemm(act16, wdn, moe, S_LEN, H_DIM, IMOE, e > 0);
  }

  // ---- shared dense MLP, accumulated into moe ----
  gemm(hs216, w_sg16, gbuf, S_LEN, IDENSE, H_DIM, 0);
  gemm(hs216, w_su16, ubuf, S_LEN, IDENSE, H_DIM, 0);
  {
    int tot = S_LEN * IDENSE;
    silu_mul_kernel<<<(tot + 255) / 256, blk, 0, stream>>>(
        gbuf, IDENSE, ubuf, IDENSE, gact16, IDENSE, nullptr, 0, IDENSE, tot);
  }
  gemm(gact16, w_sd16, moe, S_LEN, H_DIM, IDENSE, 1);

  // ---- final residual + norm ----
  add_rmsnorm2048_kernel<<<S_LEN, blk, 0, stream>>>(hs2, moe, pm_norm_w,
                                                    out, nullptr);
}
